// DiscreteConditionalEntropyModel_66769561583990
// MI455X (gfx1250) — matrix operand order repaired and checked by reading, not compiled, not run on hardware
//
#include <hip/hip_runtime.h>

// ---------------------------------------------------------------------------
// Problem sizes (fixed by the reference): B=4, N=16384, C=256, D=2048
//   M = B*N = 65536 rows
// Outputs (flat, in order): log_pmf [M*C], params_quantized [M*C], param_bit [1]
// ---------------------------------------------------------------------------

typedef __attribute__((ext_vector_type(16))) __bf16 v16bf;
typedef __attribute__((ext_vector_type(8)))  __bf16 v8bf;
typedef __attribute__((ext_vector_type(8)))  float  v8f;

#define C_DIM 256
#define D_DIM 2048
#define M_DIM 65536

// LDS tile geometry: 16 d-rows x 256 K, bf16, row pitch padded to 264 elems
// (528B -> +4 banks per row: 16-lane b128 reads tile all 64 banks, no conflict)
#define LPITCH   264
#define LMAT     (16 * LPITCH)          // one matrix (BH or BL) image
#define LBUF     (2 * LMAT)             // BH + BL
#define LTOT     (2 * LBUF)             // double buffer

static __device__ __forceinline__ __bf16 u16_to_bf(unsigned short h) {
    union { unsigned short u; __bf16 b; } x; x.u = h; return x.b;
}
static __device__ __forceinline__ float bf_to_f(__bf16 b) {
    union { unsigned short u; __bf16 b; } x; x.b = b;
    union { unsigned u; float f; } y; y.u = ((unsigned)x.u) << 16;
    return y.f;
}
static __device__ __forceinline__ __bf16 f_to_bf(float f) {   // round-to-nearest-even
    union { float f; unsigned u; } x; x.f = f;
    unsigned r = (x.u + 0x7FFFu + ((x.u >> 16) & 1u)) >> 16;
    return u16_to_bf((unsigned short)r);
}

// ---------------------------------------------------------------------------
// Kernel 1: log_softmax over logits (D=2048), single block of 256 threads.
// ---------------------------------------------------------------------------
__global__ void prep_logits_kernel(const float* __restrict__ logits,
                                   float* __restrict__ llp) {
    __shared__ float red[256];
    const int t = threadIdx.x;
    float m = -3.0e38f;
    for (int d = t; d < D_DIM; d += 256) m = fmaxf(m, logits[d]);
    red[t] = m; __syncthreads();
    for (int k = 128; k > 0; k >>= 1) {
        if (t < k) red[t] = fmaxf(red[t], red[t + k]);
        __syncthreads();
    }
    const float mx = red[0];
    __syncthreads();
    float s = 0.0f;
    for (int d = t; d < D_DIM; d += 256) s += __expf(logits[d] - mx);
    red[t] = s; __syncthreads();
    for (int k = 128; k > 0; k >>= 1) {
        if (t < k) red[t] += red[t + k];
        __syncthreads();
    }
    const float lse = mx + __logf(red[0]);
    for (int d = t; d < D_DIM; d += 256) llp[d] = logits[d] - lse;
}

// ---------------------------------------------------------------------------
// Kernel 2: split table into bf16 hi/lo copies, compute 0.5*||w_d||^2.
// One block (256 threads) per codebook row d.
// ---------------------------------------------------------------------------
__global__ __launch_bounds__(256) void prep_table_kernel(
    const float* __restrict__ W,
    __bf16* __restrict__ tHi, __bf16* __restrict__ tLo,
    float* __restrict__ wsqh) {
    __shared__ float red[256];
    const int d = blockIdx.x, c = threadIdx.x;
    const float w = W[(size_t)d * C_DIM + c];
    const __bf16 h = f_to_bf(w);
    tHi[(size_t)d * C_DIM + c] = h;
    tLo[(size_t)d * C_DIM + c] = f_to_bf(w - bf_to_f(h));
    red[c] = w * w; __syncthreads();
    for (int k = 128; k > 0; k >>= 1) {
        if (c < k) red[c] += red[c + k];
        __syncthreads();
    }
    if (c == 0) wsqh[d] = 0.5f * red[0];
}

// ---------------------------------------------------------------------------
// Kernel 3: fused GEMM + running argmax with async-LDS double buffering.
// Block = 256 threads = 8 waves in barrier lockstep over the 128 d-tiles.
// Per tile: one 16KB global->LDS copy for the whole block (4 async
// instructions per wave, ASYNCcnt-tracked, two tiles in flight), then each
// wave reads its fragments with conflict-free ds_load_b128 and issues
// 24 v_wmma_f32_16x16x32_bf16 (hi*hi + lo*hi + hi*lo, 3 independent chains).
// A fragments follow the documented 16-bit A 16x32 layout:
//   lanes 0-15 : M=lane,  elems 0..7 = K 0..7,  elems 8..15 = K 16..23
//   lanes16-31 : M=l-16,  elems 0..7 = K 8..15, elems 8..15 = K 24..31
// B (32x16) : lane = N, same K order as A (lanes 0-15: K 0..7,16..23; lanes 16-31: K 8..15,24..31).
// C/D (16x16 f32): VGPR i = row i (lanes 0-15) / row 8+i (lanes 16-31).
// ---------------------------------------------------------------------------
__global__ __launch_bounds__(256) void gemm_argmax_kernel(
    const float* __restrict__ params,
    const __bf16* __restrict__ tHi, const __bf16* __restrict__ tLo,
    const float* __restrict__ wsqh,
    int* __restrict__ index) {
    __shared__ __align__(16) __bf16 smem[LTOT];

    const int tid  = threadIdx.x;
    const int lane = tid & 31;
    const int wave = tid >> 5;
    const int half = lane >> 4;
    const int l16  = lane & 15;
    const int r0   = blockIdx.x * 128 + wave * 16;

    // ---- staging geometry: thread copies 64B of one tile row ----
    const int smat = tid >> 7;                 // 0 = BH, 1 = BL
    const int srow = (tid & 127) >> 3;         // d-row within tile
    const int scol = (tid & 7) * 32;           // element col (64B chunk)
    const __bf16* sg0  = (smat ? tLo : tHi) + (size_t)srow * C_DIM + scol;
    __bf16* sdst = &smem[0] + smat * LMAT + srow * LPITCH + scol;
    const unsigned ldsoff0 = (unsigned)(size_t)sdst;             // buf 0
    const unsigned ldsoff1 = ldsoff0 + LBUF * 2;                 // buf 1 (bytes)

    // issue one tile fill (4 x b128 per wave, lands on ASYNCcnt)
    #define FILL_TILE(jj, ldsoff)                                               \
        do {                                                                    \
            const __bf16* _s = sg0 + (size_t)(jj) * 16 * C_DIM;                 \
            asm volatile("global_load_async_to_lds_b128 %0, %1, off"            \
                         :: "v"(ldsoff), "v"(_s) : "memory");                   \
            asm volatile("global_load_async_to_lds_b128 %0, %1, off offset:16"  \
                         :: "v"(ldsoff), "v"(_s) : "memory");                   \
            asm volatile("global_load_async_to_lds_b128 %0, %1, off offset:32"  \
                         :: "v"(ldsoff), "v"(_s) : "memory");                   \
            asm volatile("global_load_async_to_lds_b128 %0, %1, off offset:48"  \
                         :: "v"(ldsoff), "v"(_s) : "memory");                   \
        } while (0)

    // ---- load + clamp + hi/lo-split this wave's 16x256 A panel ----
    v16bf aH[8], aL[8];
    const float* prow = params + (size_t)(r0 + l16) * C_DIM;
    #pragma unroll
    for (int s = 0; s < 8; ++s) {
        const int b1 = s * 32 + half * 8;          // first 8-K run
        const float4* q0 = reinterpret_cast<const float4*>(prow + b1);
        const float4* q1 = reinterpret_cast<const float4*>(prow + b1 + 16);
        float4 x0 = q0[0], x1 = q0[1], y0 = q1[0], y1 = q1[1];
        const float f[16] = { x0.x, x0.y, x0.z, x0.w,  x1.x, x1.y, x1.z, x1.w,
                              y0.x, y0.y, y0.z, y0.w,  y1.x, y1.y, y1.z, y1.w };
        #pragma unroll
        for (int e = 0; e < 16; ++e) {
            const float p  = fminf(1.0f, fmaxf(-1.0f, f[e]));
            const __bf16 h = f_to_bf(p);
            aH[s][e] = h;
            aL[s][e] = f_to_bf(p - bf_to_f(h));
        }
    }

    // ---- async pipeline prologue: tiles 0 and 1 in flight ----
    FILL_TILE(0, ldsoff0);
    FILL_TILE(1, ldsoff1);
    asm volatile("s_wait_asynccnt 0x4" ::: "memory");   // tile 0 complete
    __syncthreads();                                    // ...and published

    float bestV[8];
    int   bestI[8];
    #pragma unroll
    for (int i = 0; i < 8; ++i) { bestV[i] = -3.0e38f; bestI[i] = 0; }

    for (int j = 0; j < D_DIM / 16; ++j) {
        const int d = j * 16 + l16;                // this lane's column (N index)
        const __bf16* pBH = &smem[0] + (j & 1) * LBUF + l16 * LPITCH + half * 8;
        const __bf16* pBL = pBH + LMAT;

        v8f accHH = {}, accLH = {}, accHL = {};    // independent XDL chains
        #pragma unroll
        for (int s = 0; s < 8; ++s) {
            union { v16bf v; v8bf h[2]; } BH, BL;
            BH.h[0] = *reinterpret_cast<const v8bf*>(pBH + s * 32);
            BH.h[1] = *reinterpret_cast<const v8bf*>(pBH + s * 32 + 16);
            BL.h[0] = *reinterpret_cast<const v8bf*>(pBL + s * 32);
            BL.h[1] = *reinterpret_cast<const v8bf*>(pBL + s * 32 + 16);
            accHH = __builtin_amdgcn_wmma_f32_16x16x32_bf16(
                        false, aH[s], false, BH.v, (short)0, accHH, false, false);
            accLH = __builtin_amdgcn_wmma_f32_16x16x32_bf16(
                        false, aL[s], false, BH.v, (short)0, accLH, false, false);
            accHL = __builtin_amdgcn_wmma_f32_16x16x32_bf16(
                        false, aH[s], false, BL.v, (short)0, accHL, false, false);
        }

        const float wq = wsqh[d];
        #pragma unroll
        for (int i = 0; i < 8; ++i) {
            const float v = (accHH[i] + accLH[i] + accHL[i]) - wq;  // p.w - 0.5||w||^2
            if (v > bestV[i] || (v == bestV[i] && d < bestI[i])) {
                bestV[i] = v; bestI[i] = d;
            }
        }

        __syncthreads();                           // all waves done reading buf[j&1]
        const int jn = (j + 2 < D_DIM / 16) ? (j + 2) : (D_DIM / 16 - 1);
        FILL_TILE(jn, (j & 1) ? ldsoff1 : ldsoff0);                 // refill with tile j+2
        asm volatile("s_wait_asynccnt 0x4" ::: "memory");           // tile j+1 complete
        __syncthreads();                                            // ...and published
    }

    // ---- reduce (val,idx) across the 16 lanes of each half ----
    #pragma unroll
    for (int i = 0; i < 8; ++i) {
        float v = bestV[i]; int id = bestI[i];
        #pragma unroll
        for (int off = 8; off >= 1; off >>= 1) {
            const float ov = __shfl_xor(v, off, 32);
            const int   oi = __shfl_xor(id, off, 32);
            if (ov > v || (ov == v && oi < id)) { v = ov; id = oi; }
        }
        if (l16 == 0) index[r0 + half * 8 + i] = id;   // lane0 rows 0..7, lane16 rows 8..15
    }
    #undef FILL_TILE
}

// ---------------------------------------------------------------------------
// Kernel 4: gather codebook row + log_softmax per row; one wave per row.
// ---------------------------------------------------------------------------
__global__ __launch_bounds__(256) void output_kernel(
    const float* __restrict__ table, const int* __restrict__ index,
    float* __restrict__ out) {
    const int lane = threadIdx.x & 31;
    const int wave = threadIdx.x >> 5;
    const int row  = blockIdx.x * 8 + wave;
    const int idx  = index[row];
    const float* q = table + (size_t)idx * C_DIM + lane * 8;
    const float4* q4 = reinterpret_cast<const float4*>(q);
    float4 a = q4[0], b = q4[1];
    const float v[8] = { a.x, a.y, a.z, a.w, b.x, b.y, b.z, b.w };

    float m = v[0];
    #pragma unroll
    for (int t = 1; t < 8; ++t) m = fmaxf(m, v[t]);
    #pragma unroll
    for (int off = 16; off >= 1; off >>= 1) m = fmaxf(m, __shfl_xor(m, off, 32));

    float s = 0.0f;
    #pragma unroll
    for (int t = 0; t < 8; ++t) s += __expf(v[t] - m);
    #pragma unroll
    for (int off = 16; off >= 1; off >>= 1) s += __shfl_xor(s, off, 32);
    const float lse = m + __logf(s);

    const size_t base = (size_t)row * C_DIM + lane * 8;
    float4 o0 = { v[0] - lse, v[1] - lse, v[2] - lse, v[3] - lse };
    float4 o1 = { v[4] - lse, v[5] - lse, v[6] - lse, v[7] - lse };
    float4* lp = reinterpret_cast<float4*>(out + base);
    lp[0] = o0; lp[1] = o1;
    float4* pq = reinterpret_cast<float4*>(out + (size_t)M_DIM * C_DIM + base);
    pq[0] = a; pq[1] = b;
}

// ---------------------------------------------------------------------------
// Kernel 5: deterministic scalar reduction for param_bit (fixed order, no
// float atomics -> bit-identical across graph replays).
// ---------------------------------------------------------------------------
__global__ void bit_kernel(const float* __restrict__ llp,
                           const int* __restrict__ index,
                           float* __restrict__ out_scalar) {
    __shared__ float red[256];
    const int t = threadIdx.x;
    float s = 0.0f;
    for (int r = t; r < M_DIM; r += 256) s += llp[index[r]];
    red[t] = s; __syncthreads();
    for (int k = 128; k > 0; k >>= 1) {
        if (t < k) red[t] += red[t + k];
        __syncthreads();
    }
    if (t == 0) *out_scalar = red[0] * (-1.44269504088896340736f);  // / (-ln2)
}

// ---------------------------------------------------------------------------
extern "C" void kernel_launch(void* const* d_in, const int* in_sizes, int n_in,
                              void* d_out, int out_size, void* d_ws, size_t ws_size,
                              hipStream_t stream) {
    const float* params = (const float*)d_in[0];   // (4,16384,256) f32
    const float* table  = (const float*)d_in[1];   // (2048,256)    f32
    const float* logits = (const float*)d_in[2];   // (2048,)       f32
    float* out = (float*)d_out;                    // 2*M*C + 1 floats

    // workspace layout (~2.27 MB)
    float*  wsqh = (float*)d_ws;                           // D
    float*  llp  = wsqh + D_DIM;                           // D
    __bf16* tHi  = (__bf16*)(llp + D_DIM);                 // D*C
    __bf16* tLo  = tHi + (size_t)D_DIM * C_DIM;            // D*C
    int*    idx  = (int*)(tLo + (size_t)D_DIM * C_DIM);    // M

    prep_logits_kernel<<<1, 256, 0, stream>>>(logits, llp);
    prep_table_kernel<<<D_DIM, 256, 0, stream>>>(table, tHi, tLo, wsqh);
    gemm_argmax_kernel<<<M_DIM / 128, 256, 0, stream>>>(params, tHi, tLo, wsqh, idx);
    output_kernel<<<M_DIM / 8, 256, 0, stream>>>(table, idx, out);
    bit_kernel<<<1, 256, 0, stream>>>(llp, idx, out + (size_t)2 * M_DIM * C_DIM);
}
